// EGraphSAGE_83485574300032
// MI455X (gfx1250) — matrix operand order repaired and checked by reading, not compiled, not run on hardware
//
#include <hip/hip_runtime.h>
#include <hip/hip_fp16.h>

typedef __attribute__((ext_vector_type(16))) _Float16 v16h;
typedef __attribute__((ext_vector_type(8)))  _Float16 v8h;
typedef __attribute__((ext_vector_type(8)))  float    v8f;
typedef __attribute__((ext_vector_type(4)))  float    v4f;

#define BN_EPS 1e-5f

// ---------------------------------------------------------------------------
// WMMA helper (CDNA5 gfx1250, wave32). D = A(16x32 f16) * B(32x16 f16) + C(f32)
// ---------------------------------------------------------------------------
static __device__ __forceinline__ v8f wmma_f16(v16h a, v16h b, v8f c) {
  return __builtin_amdgcn_wmma_f32_16x16x32_f16(
      /*neg_a=*/false, a, /*neg_b=*/false, b,
      /*c_mod=*/(short)0, c, /*reuse_a=*/false, /*reuse_b=*/false);
}

// A-fragment from an f32 row (row = per-lane pointer to this lane's M-row).
// ISA 7.12.2 (16-bit A 16x32): lane m = lane&15, g = lane>>4;
// halves j=0..7 -> k = kt*32 + 8g + j ; j=8..15 -> k = kt*32 + 16 + 8g + (j-8)
static __device__ __forceinline__ v16h a_frag_f32(const float* row, int kt, float scale) {
  int lane = threadIdx.x & 31;
  int g = (lane >> 4) & 1;
  const float* p0 = row + kt * 32 + 8 * g;
  const float* p1 = row + kt * 32 + 16 + 8 * g;
  v4f a0 = *(const v4f*)(p0);
  v4f a1 = *(const v4f*)(p0 + 4);
  v4f a2 = *(const v4f*)(p1);
  v4f a3 = *(const v4f*)(p1 + 4);
  v16h f;
#pragma unroll
  for (int i = 0; i < 4; ++i) {
    f[i]      = (_Float16)(a0[i] * scale);
    f[4 + i]  = (_Float16)(a1[i] * scale);
    f[8 + i]  = (_Float16)(a2[i] * scale);
    f[12 + i] = (_Float16)(a3[i] * scale);
  }
  return f;
}

// A-fragment from an f16 row (global h1 table or per-wave LDS tile).
static __device__ __forceinline__ v16h a_frag_f16(const _Float16* row, int kt) {
  int lane = threadIdx.x & 31;
  int g = (lane >> 4) & 1;
  v8h lo = *(const v8h*)(row + kt * 32 + 8 * g);
  v8h hi = *(const v8h*)(row + kt * 32 + 16 + 8 * g);
  v16h f;
#pragma unroll
  for (int i = 0; i < 8; ++i) { f[i] = lo[i]; f[8 + i] = hi[i]; }
  return f;
}

// B-fragments pre-packed per-lane-contiguous (32 B/lane) -> 2x global_load_b128
static __device__ __forceinline__ v16h b_frag(const _Float16* fr, int kt, int nt) {
  int lane = threadIdx.x & 31;
  return *(const v16h*)(fr + (unsigned)((kt * 8 + nt) * 512 + lane * 16));
}

// ---------------------------------------------------------------------------
// Pack W[Nout][K] (row-major f32) into f16 B-fragments. B[k][n] = W[n][k].
// Fragment f = kt*8 + nt (Nout always 128 -> 8 n-tiles). One wave per fragment.
// B layout: lane n = lane&15, g = lane>>4, j=0..7 -> k = kt*32+8g+j ; j=8..15 -> k = kt*32+16+8g+(j-8).
// ---------------------------------------------------------------------------
__global__ void prep_weight_frags(const float* __restrict__ W, int krow, int col_off,
                                  _Float16* __restrict__ dst) {
  int f = blockIdx.x;
  int lane = threadIdx.x & 31;
  int nt = f & 7, kt = f >> 3;
  int g = (lane >> 4) & 1, n = lane & 15;
  const float* src = W + (size_t)(nt * 16 + n) * krow + col_off + kt * 32;
  v16h out;
#pragma unroll
  for (int j = 0; j < 16; ++j) out[j] = (_Float16)src[(j < 8) ? (8 * g + j) : (16 + 8 * g + (j - 8))];
  *(v16h*)(dst + (unsigned)(f * 512 + lane * 16)) = out;
}

// ---------------------------------------------------------------------------
// wcomb = Wn(128x128) @ We(128x64)  -> [128][64] f32.  (1 MFLOP, one-off)
// Exploits linearity: mean(edge_attr@We^T)@Wn^T == mean(edge_attr)@(Wn@We)^T
// ---------------------------------------------------------------------------
__global__ void combine_weights(const float* __restrict__ wn,
                                const float* __restrict__ we,
                                float* __restrict__ out) {
  int idx = blockIdx.x * blockDim.x + threadIdx.x;  // 128*64 = 8192
  int i = idx >> 6, j = idx & 63;
  float s = 0.0f;
#pragma unroll 8
  for (int k = 0; k < 128; ++k) s += wn[i * 128 + k] * we[k * 64 + j];
  out[idx] = s;
}

// ---------------------------------------------------------------------------
// Degree histogram over destination nodes.
// ---------------------------------------------------------------------------
__global__ void deg_kernel(const int* __restrict__ dst, float* __restrict__ deg, int E) {
  int i = blockIdx.x * blockDim.x + threadIdx.x;
  if (i < E) atomicAdd(&deg[dst[i]], 1.0f);
}

// ---------------------------------------------------------------------------
// Scatter-sum of RAW edge_attr onto dst nodes (xsum[v] = sum of 64-wide rows).
// One thread per float4 quad: b128 coalesced read, 4 f32 atomics with 32-bit
// offsets against a uniform base (saddr+voffset form). 102M lane-atomics total
// -- 4x fewer than scattering the two 128-wide projections.
// ---------------------------------------------------------------------------
__global__ void edge_sum(const float* __restrict__ eattr,
                         const int* __restrict__ dst,
                         float* __restrict__ xsum, int numQuads) {
  int q = blockIdx.x * blockDim.x + threadIdx.x;
  if (q >= numQuads) return;
  __builtin_prefetch(eattr + (size_t)q * 4 + 16384, 0, 1);  // stream-ahead
  v4f v = *(const v4f*)(eattr + (size_t)q * 4);
  int e = q >> 4;
  unsigned off = (unsigned)dst[e] * 64u + (unsigned)(q & 15) * 4u;
  atomicAdd(&xsum[off + 0], v[0]);
  atomicAdd(&xsum[off + 1], v[1]);
  atomicAdd(&xsum[off + 2], v[2]);
  atomicAdd(&xsum[off + 3], v[3]);
}

// ---------------------------------------------------------------------------
// Fused node pipeline: both GraphSAGE layers in one kernel (layer-1 depends
// only on the node's own h0 row). Layer-0 D-tile is transposed D->A layout
// through a per-wave LDS tile (ds_store_b16 / ds_load_b128), so h0 never
// touches HBM. m = xsum/deg is shared by both layers via combined weights.
// ---------------------------------------------------------------------------
__global__ void node_fused(const float* __restrict__ x,
                           const float* __restrict__ xsum,
                           const float* __restrict__ deg,
                           const _Float16* __restrict__ fws0,   // w_self_0  (K=64)
                           const _Float16* __restrict__ fc0,    // wn0@we0   (K=64)
                           const _Float16* __restrict__ fws1,   // w_self_1  (K=128)
                           const _Float16* __restrict__ fc1,    // wn1@we1   (K=64)
                           const float* __restrict__ b0,
                           const float* __restrict__ g0, const float* __restrict__ be0,
                           const float* __restrict__ m0, const float* __restrict__ v0,
                           const float* __restrict__ b1,
                           const float* __restrict__ g1, const float* __restrict__ be1,
                           const float* __restrict__ m1, const float* __restrict__ v1,
                           _Float16* __restrict__ h1, int numTiles) {
  __shared__ _Float16 lds_tile[8][16 * 128];  // 4 KB per wave, 32 KB total
  int wave = threadIdx.x >> 5;
  int lane = threadIdx.x & 31;
  int g = (lane >> 4) & 1, n = lane & 15;
  int t = blockIdx.x * (blockDim.x >> 5) + wave;
  if (t >= numTiles) return;   // no block-wide barriers used -> safe
  int rbase = t * 16;
  int arow = rbase + n;
  float sc = 1.0f / fmaxf(deg[arow], 1.0f);   // mean aggregation
  const float* xrow = x + (size_t)arow * 64;
  const float* mrow = xsum + (size_t)arow * 64;
  v16h ax0 = a_frag_f32(xrow, 0, 1.0f);
  v16h ax1 = a_frag_f32(xrow, 1, 1.0f);
  v16h am0 = a_frag_f32(mrow, 0, sc);
  v16h am1 = a_frag_f32(mrow, 1, sc);
  _Float16* tp = &lds_tile[wave][0];

  // ----- layer 0: h0 = relu(bn0(x@Ws0^T + m@C0^T + b0)) -> LDS (f16) -----
  for (int nt = 0; nt < 8; ++nt) {
    v8f acc = {};
    acc = wmma_f16(ax0, b_frag(fws0, 0, nt), acc);
    acc = wmma_f16(ax1, b_frag(fws0, 1, nt), acc);
    acc = wmma_f16(am0, b_frag(fc0, 0, nt), acc);
    acc = wmma_f16(am1, b_frag(fc0, 1, nt), acc);
    int c = nt * 16 + n;
    float bb = b0[c];
    float scale = g0[c] * rsqrtf(v0[c] + BN_EPS);
    float shift = be0[c] - m0[c] * scale;
#pragma unroll
    for (int r = 0; r < 8; ++r) {
      float hv = fmaxf((acc[r] + bb) * scale + shift, 0.0f);
      tp[(r + 8 * g) * 128 + c] = (_Float16)hv;   // D-layout -> row-major LDS
    }
  }

  // ----- layer 1: h1 = relu(bn1(h0@Ws1^T + m@C1^T + b1)) -> global f16 -----
  v16h ah[4];
#pragma unroll
  for (int kt = 0; kt < 4; ++kt) ah[kt] = a_frag_f16(tp + n * 128, kt);
  for (int nt = 0; nt < 8; ++nt) {
    v8f acc = {};
#pragma unroll
    for (int kt = 0; kt < 4; ++kt) acc = wmma_f16(ah[kt], b_frag(fws1, kt, nt), acc);
    acc = wmma_f16(am0, b_frag(fc1, 0, nt), acc);
    acc = wmma_f16(am1, b_frag(fc1, 1, nt), acc);
    int c = nt * 16 + n;
    float bb = b1[c];
    float scale = g1[c] * rsqrtf(v1[c] + BN_EPS);
    float shift = be1[c] - m1[c] * scale;
#pragma unroll
    for (int r = 0; r < 8; ++r) {
      float hv = fmaxf((acc[r] + bb) * scale + shift, 0.0f);
      h1[(unsigned)(rbase + r + 8 * g) * 128u + c] = (_Float16)hv;
    }
  }
}

// ---------------------------------------------------------------------------
// Edge classifier: hid = relu([h1[src], h1[dst]] @ wc1^T + bc1); out = hid@wc2^T+bc2.
// K=256 split: A_src vs wc1[:, :128], A_dst vs wc1[:, 128:]. Final 128->2 via
// per-lane partials + wave32 xor-shuffle reduction (masks 1/2/4/8 stay in-group).
// ---------------------------------------------------------------------------
__global__ void edge_classifier(const _Float16* __restrict__ h1,
                                const int* __restrict__ srcIdx,
                                const int* __restrict__ dstIdx,
                                const _Float16* __restrict__ fra,
                                const _Float16* __restrict__ frb,
                                const float* __restrict__ bc1,
                                const float* __restrict__ wc2,
                                const float* __restrict__ bc2,
                                float* __restrict__ out, int numTiles) {
  int wave = threadIdx.x >> 5;
  int lane = threadIdx.x & 31;
  int g = (lane >> 4) & 1, n = lane & 15;
  int t = blockIdx.x * (blockDim.x >> 5) + wave;
  if (t >= numTiles) return;
  int ebase = t * 16;
  const _Float16* srow = h1 + (size_t)((unsigned)srcIdx[ebase + n] * 128u);
  const _Float16* drow = h1 + (size_t)((unsigned)dstIdx[ebase + n] * 128u);
  v16h as[4], ad[4];
#pragma unroll
  for (int kt = 0; kt < 4; ++kt) {
    as[kt] = a_frag_f16(srow, kt);
    ad[kt] = a_frag_f16(drow, kt);
  }
  float p0[8], p1[8];
#pragma unroll
  for (int r = 0; r < 8; ++r) { p0[r] = 0.0f; p1[r] = 0.0f; }
  for (int nt = 0; nt < 8; ++nt) {
    v8f acc = {};
#pragma unroll
    for (int kt = 0; kt < 4; ++kt) acc = wmma_f16(as[kt], b_frag(fra, kt, nt), acc);
#pragma unroll
    for (int kt = 0; kt < 4; ++kt) acc = wmma_f16(ad[kt], b_frag(frb, kt, nt), acc);
    int c = nt * 16 + n;
    float b = bc1[c];
    float w0 = wc2[c];        // wc2[0][c]
    float w1 = wc2[128 + c];  // wc2[1][c]
#pragma unroll
    for (int r = 0; r < 8; ++r) {
      float hv = fmaxf(acc[r] + b, 0.0f);
      p0[r] += hv * w0;
      p1[r] += hv * w1;
    }
  }
#pragma unroll
  for (int off = 1; off < 16; off <<= 1) {
#pragma unroll
    for (int r = 0; r < 8; ++r) {
      p0[r] += __shfl_xor(p0[r], off, 32);
      p1[r] += __shfl_xor(p1[r], off, 32);
    }
  }
  if (n == 0) {
    float q0 = bc2[0], q1 = bc2[1];
#pragma unroll
    for (int r = 0; r < 8; ++r) {
      unsigned e = (unsigned)(ebase + r + 8 * g);
      out[e * 2u + 0] = p0[r] + q0;
      out[e * 2u + 1] = p1[r] + q1;
    }
  }
}

// ---------------------------------------------------------------------------
// Launch
// ---------------------------------------------------------------------------
extern "C" void kernel_launch(void* const* d_in, const int* in_sizes, int n_in,
                              void* d_out, int out_size, void* d_ws, size_t ws_size,
                              hipStream_t stream) {
  const float* x        = (const float*)d_in[0];
  const int*   eidx     = (const int*)d_in[1];
  const float* eattr    = (const float*)d_in[2];
  const int*   elidx    = (const int*)d_in[3];
  const float* w_self_0 = (const float*)d_in[4];
  const float* w_neigh_0= (const float*)d_in[5];
  const float* w_edge_0 = (const float*)d_in[6];
  const float* b_0      = (const float*)d_in[7];
  const float* bn_g_0   = (const float*)d_in[8];
  const float* bn_b_0   = (const float*)d_in[9];
  const float* bn_m_0   = (const float*)d_in[10];
  const float* bn_v_0   = (const float*)d_in[11];
  const float* w_self_1 = (const float*)d_in[12];
  const float* w_neigh_1= (const float*)d_in[13];
  const float* w_edge_1 = (const float*)d_in[14];
  const float* b_1      = (const float*)d_in[15];
  const float* bn_g_1   = (const float*)d_in[16];
  const float* bn_b_1   = (const float*)d_in[17];
  const float* bn_m_1   = (const float*)d_in[18];
  const float* bn_v_1   = (const float*)d_in[19];
  const float* wc1      = (const float*)d_in[20];
  const float* bc1      = (const float*)d_in[21];
  const float* wc2      = (const float*)d_in[22];
  const float* bc2      = (const float*)d_in[23];
  float* outp = (float*)d_out;

  const int N = in_sizes[0] / 64;   // 50000
  const int E = in_sizes[1] / 2;    // 1600000
  const int* edge_dst = eidx + E;   // edge_index[1]
  const int* lbl_src  = elidx;      // edge_label_index[0]
  const int* lbl_dst  = elidx + E;  // edge_label_index[1]

  // ---- workspace layout (256B aligned sections) ----
  char* ws = (char*)d_ws;
  size_t off = 0;
  auto take = [&](size_t bytes) -> char* {
    char* p = ws + off;
    off = (off + bytes + 255) & ~(size_t)255;
    return p;
  };
  float*    xsum = (float*)take((size_t)N * 64 * 4);
  float*    deg  = (float*)take((size_t)N * 4);
  size_t zeroBytes = off;                    // xsum + deg start at offset 0
  _Float16* h1   = (_Float16*)take((size_t)N * 128 * 2);
  float*    wcb0 = (float*)take(128 * 64 * 4);
  float*    wcb1 = (float*)take(128 * 64 * 4);
  _Float16* fr_ws0  = (_Float16*)take(16 * 1024);
  _Float16* fr_c0   = (_Float16*)take(16 * 1024);
  _Float16* fr_ws1  = (_Float16*)take(32 * 1024);
  _Float16* fr_c1   = (_Float16*)take(16 * 1024);
  _Float16* fr_wc1a = (_Float16*)take(32 * 1024);
  _Float16* fr_wc1b = (_Float16*)take(32 * 1024);
  if (off > ws_size) return;  // scratch too small; bail deterministically

  hipMemsetAsync(d_ws, 0, zeroBytes, stream);

  // ---- one-off weight preparation ----
  combine_weights<<<32, 256, 0, stream>>>(w_neigh_0, w_edge_0, wcb0);
  combine_weights<<<32, 256, 0, stream>>>(w_neigh_1, w_edge_1, wcb1);
  prep_weight_frags<<<16, 32, 0, stream>>>(w_self_0,  64,   0, fr_ws0);
  prep_weight_frags<<<16, 32, 0, stream>>>(wcb0,      64,   0, fr_c0);
  prep_weight_frags<<<32, 32, 0, stream>>>(w_self_1,  128,  0, fr_ws1);
  prep_weight_frags<<<16, 32, 0, stream>>>(wcb1,      64,   0, fr_c1);
  prep_weight_frags<<<32, 32, 0, stream>>>(wc1,       256,  0, fr_wc1a);
  prep_weight_frags<<<32, 32, 0, stream>>>(wc1,       256, 128, fr_wc1b);

  // ---- degree histogram + raw edge_attr scatter-sum ----
  deg_kernel<<<(E + 255) / 256, 256, 0, stream>>>(edge_dst, deg, E);
  int numQuads = E * 16;  // E rows of 64 floats = 16 float4 each
  edge_sum<<<(numQuads + 255) / 256, 256, 0, stream>>>(eattr, edge_dst, xsum, numQuads);

  // ---- fused two-layer node update ----
  int tilesN = N / 16;
  int blocksN = (tilesN + 7) / 8;
  node_fused<<<blocksN, 256, 0, stream>>>(x, xsum, deg,
                                          fr_ws0, fr_c0, fr_ws1, fr_c1,
                                          b_0, bn_g_0, bn_b_0, bn_m_0, bn_v_0,
                                          b_1, bn_g_1, bn_b_1, bn_m_1, bn_v_1,
                                          h1, tilesN);

  // ---- edge classifier ----
  int tilesE = E / 16;
  int blocksC = (tilesE + 7) / 8;
  edge_classifier<<<blocksC, 256, 0, stream>>>(h1, lbl_src, lbl_dst,
                                               fr_wc1a, fr_wc1b,
                                               bc1, wc2, bc2, outp, tilesE);
  (void)n_in; (void)out_size;
}
